// S6GatedAttention_49520972923223
// MI455X (gfx1250) — matrix operand order repaired and checked by reading, not compiled, not run on hardware
//
#include <hip/hip_runtime.h>
#include <hip/hip_bf16.h>
#include <math.h>
#include <stdint.h>

typedef __bf16 bf16;
typedef __attribute__((ext_vector_type(16))) __bf16 v16bf;
typedef __attribute__((ext_vector_type(8)))  float  v8f;

#define EMBED_DIM 1024
#define ZDIM      128
#define CHUNK     2048
#define DSTATE    16
#define DT_RANK   64
#define BATCH     2
#define SEQ       4096
#define ROWS      (BATCH*SEQ)          /* 8192 token rows */
#define XDBL_N    (DT_RANK + 2*DSTATE) /* 96 */

/* ------------------------------------------------------------------ */
/* Generic bf16 WMMA GEMM:  C[MxN] = act( A[MxK] * W^T + bias )        */
/* Tile: 128x128x32, 256 threads = 8 waves, wave tile 64x32 (8 wmma).  */
/* Contract at all call sites: M % 128 == 0, K % 32 == 0.              */
/* WSRC: 0 = fp32 weights (converted), 1 = bf16 weights.               */
/* WTRANS: 0 = W is N x K (W[n*ldw+k]), 1 = W is K x N (W[k*ldw+n]).   */
/* ACT: 0 none, 1 silu, 2 sigmoid, 3 softplus (compile-time).          */
/* A tile + bf16 B tile staged with GLOBAL_LOAD_ASYNC_TO_LDS_B128.     */
/* ------------------------------------------------------------------ */
#define BM 128
#define BN 128
#define BK 32
#define LDSK (BK + 8)

__device__ __forceinline__ void async_b128_to_lds(void* lds_dst, const void* gsrc)
{
    /* low 32 bits of a flat shared address are the LDS byte offset    */
    unsigned lds_off = (unsigned)(uintptr_t)lds_dst;
    uint64_t gaddr   = (uint64_t)(uintptr_t)gsrc;
    asm volatile("global_load_async_to_lds_b128 %0, %1, off"
                 :: "v"(lds_off), "v"(gaddr) : "memory");
}
__device__ __forceinline__ void wait_asynccnt0()
{
    asm volatile("s_wait_asynccnt 0x0" ::: "memory");
}

template <int WSRC, int WTRANS, int ACT>
__global__ __launch_bounds__(256)
void gemm_wmma(const bf16* __restrict__ A, int lda,
               const void* __restrict__ Wp, int ldw,
               const float* __restrict__ bias,
               float* __restrict__ Cf, bf16* __restrict__ Cb, int ldc,
               int M, int N, int K)
{
    const float* Wf = (const float*)Wp;
    const bf16*  Wb = (const bf16*)Wp;

    __shared__ __attribute__((aligned(16))) bf16 As[BM][LDSK];
    __shared__ __attribute__((aligned(16))) bf16 Bs[BN][LDSK];

    const int tid   = threadIdx.x;
    const int lane  = tid & 31;
    const int wave  = tid >> 5;
    const int waveM = wave >> 2;   /* 0..1 : 64 rows each  */
    const int waveN = wave & 3;    /* 0..3 : 32 cols each  */
    const int blockN = blockIdx.x * BN;
    const int blockM = blockIdx.y * BM;

    /* per-thread staging coordinates */
    const int arow = tid >> 1;            /* A/B row inside tile  */
    const int akc  = (tid & 1) << 4;      /* 0 or 16              */
    const int tkr  = tid >> 3;            /* 0..31  (WTRANS path) */
    const int tn0  = (tid & 7) << 4;      /* 0..112 (WTRANS path) */

    v8f acc[4][2];
    for (int i = 0; i < 4; ++i)
        for (int j = 0; j < 2; ++j)
            for (int e = 0; e < 8; ++e) acc[i][j][e] = 0.f;

    for (int k0 = 0; k0 < K; k0 += BK) {
        /* ---- prefetch next K-slab (global_prefetch_b8) */
        if (k0 + BK < K) {
            __builtin_prefetch(A + (size_t)(blockM + arow) * lda + (k0 + BK), 0, 3);
            if (WTRANS == 0)
                __builtin_prefetch((const char*)Wp +
                    ((size_t)(blockN + arow) * ldw + (k0 + BK)) * (WSRC ? 2 : 4), 0, 3);
        }

        /* ---- stage A tile: async memory->LDS, 2 x b128 per lane */
        {
            bf16* dst = &As[arow][akc];
            const bf16* src = A + (size_t)(blockM + arow) * lda + (k0 + akc);
            async_b128_to_lds(dst,     src);
            async_b128_to_lds(dst + 8, src + 8);
        }

        /* ---- stage B tile as Bs[n][k] */
        if (WTRANS == 0) {
            const int gN = blockN + arow;
            const int gK = k0 + akc;
            bf16* dst = &Bs[arow][akc];
            if (WSRC == 1) {
                /* bf16, N-major, all such call sites have N % 128 == 0 */
                const bf16* src = Wb + (size_t)gN * ldw + gK;
                async_b128_to_lds(dst,     src);
                async_b128_to_lds(dst + 8, src + 8);
            } else {
                /* fp32 weights: vector load, convert, packed b128 store */
                __attribute__((aligned(16))) bf16 tmp[16];
                if (gN < N) {
                    const float4* src = reinterpret_cast<const float4*>(
                        Wf + (size_t)gN * ldw + gK);
                    float4 f0 = src[0], f1 = src[1], f2 = src[2], f3 = src[3];
                    tmp[0]=(bf16)f0.x; tmp[1]=(bf16)f0.y; tmp[2]=(bf16)f0.z; tmp[3]=(bf16)f0.w;
                    tmp[4]=(bf16)f1.x; tmp[5]=(bf16)f1.y; tmp[6]=(bf16)f1.z; tmp[7]=(bf16)f1.w;
                    tmp[8]=(bf16)f2.x; tmp[9]=(bf16)f2.y; tmp[10]=(bf16)f2.z; tmp[11]=(bf16)f2.w;
                    tmp[12]=(bf16)f3.x; tmp[13]=(bf16)f3.y; tmp[14]=(bf16)f3.z; tmp[15]=(bf16)f3.w;
                } else {
                    for (int i = 0; i < 16; ++i) tmp[i] = (bf16)0.f;
                }
                *reinterpret_cast<uint4*>(dst)     = *reinterpret_cast<const uint4*>(tmp);
                *reinterpret_cast<uint4*>(dst + 8) = *reinterpret_cast<const uint4*>(tmp + 8);
            }
        } else {
            /* K-major weights (values matrix): N % 128 == 0 at call sites.
               Vector-load 16 contiguous n, scatter-transpose into LDS.   */
            const bf16* src = Wb + (size_t)(k0 + tkr) * ldw + (blockN + tn0);
            __attribute__((aligned(16))) bf16 tmp[16];
            *reinterpret_cast<uint4*>(tmp)     = *reinterpret_cast<const uint4*>(src);
            *reinterpret_cast<uint4*>(tmp + 8) = *(reinterpret_cast<const uint4*>(src) + 1);
            for (int i = 0; i < 16; ++i) Bs[tn0 + i][tkr] = tmp[i];
        }

        wait_asynccnt0();        /* our async b128s are done            */
        __syncthreads();         /* everyone's LDS writes visible       */

        /* ---- WMMA: A frag per ISA 16-bit A layout (lane<16 -> K 0-7,16-23) */
        v16bf afrag[4];
        const int kh = (lane < 16) ? 0 : 8;
        for (int tm = 0; tm < 4; ++tm) {
            const bf16* ap = &As[waveM * 64 + tm * 16 + (lane & 15)][kh];
            v16bf a;
            for (int i = 0; i < 8; ++i) a[i]     = ap[i];
            for (int i = 0; i < 8; ++i) a[8 + i] = ap[16 + i];
            afrag[tm] = a;
        }
        const int kb = (lane < 16) ? 0 : 8;
        for (int tn = 0; tn < 2; ++tn) {
            const bf16* bp = &Bs[waveN * 32 + tn * 16 + (lane & 15)][kb];
            v16bf b;
            for (int i = 0; i < 8; ++i) b[i]     = bp[i];
            for (int i = 0; i < 8; ++i) b[8 + i] = bp[16 + i];
            for (int tm = 0; tm < 4; ++tm) {
                acc[tm][tn] = __builtin_amdgcn_wmma_f32_16x16x32_bf16(
                    false, afrag[tm], false, b, (short)0, acc[tm][tn],
                    false, false);
            }
        }
        __syncthreads();
    }

    /* ---- epilogue: C/D layout lanes 0-15 M=r, lanes 16-31 M=8+r.
       Activate in place, then hoisted store loops.                    */
    const int rbase = (lane >> 4) << 3;
    const int col0  = blockN + waveN * 32 + (lane & 15);
    for (int tn = 0; tn < 2; ++tn) {
        const int gcol = col0 + tn * 16;
        const float bv = (bias && gcol < N) ? bias[gcol] : 0.f;
        for (int tm = 0; tm < 4; ++tm)
            for (int r = 0; r < 8; ++r) {
                float x = acc[tm][tn][r] + bv;
                if (ACT == 1)      x = x / (1.f + __expf(-x));              /* silu */
                else if (ACT == 2) x = 1.f / (1.f + __expf(-x));            /* sigmoid */
                else if (ACT == 3) x = (x > 20.f) ? x
                                        : __logf(1.f + __expf(x));          /* softplus */
                acc[tm][tn][r] = x;
            }
    }
    if (Cf) {
        for (int tn = 0; tn < 2; ++tn) {
            const int gcol = col0 + tn * 16;
            if (gcol >= N) continue;
            for (int tm = 0; tm < 4; ++tm) {
                const size_t row0 = (size_t)(blockM + waveM * 64 + tm * 16 + rbase);
                for (int r = 0; r < 8; ++r)
                    Cf[(row0 + r) * ldc + gcol] = acc[tm][tn][r];
            }
        }
    }
    if (Cb) {
        for (int tn = 0; tn < 2; ++tn) {
            const int gcol = col0 + tn * 16;
            if (gcol >= N) continue;
            for (int tm = 0; tm < 4; ++tm) {
                const size_t row0 = (size_t)(blockM + waveM * 64 + tm * 16 + rbase);
                for (int r = 0; r < 8; ++r)
                    Cb[(row0 + r) * ldc + gcol] = (bf16)acc[tm][tn][r];
            }
        }
    }
}

/* ------------------------------------------------------------------ */
__global__ __launch_bounds__(256)
void rmsnorm_kernel(const float* __restrict__ x, const float* __restrict__ w,
                    float* __restrict__ xn, bf16* __restrict__ xnb)
{
    __shared__ float red[256];
    const int row = blockIdx.x;
    const float* xr = x + (size_t)row * EMBED_DIM;
    float ss = 0.f;
    for (int i = threadIdx.x; i < EMBED_DIM; i += 256) { float v = xr[i]; ss += v * v; }
    red[threadIdx.x] = ss; __syncthreads();
    for (int s = 128; s > 0; s >>= 1) {
        if (threadIdx.x < s) red[threadIdx.x] += red[threadIdx.x + s];
        __syncthreads();
    }
    const float inv = rsqrtf(red[0] / (float)EMBED_DIM + 1e-5f);
    for (int i = threadIdx.x; i < EMBED_DIM; i += 256) {
        float v = xr[i] * inv * w[i];
        xn [(size_t)row * EMBED_DIM + i] = v;
        xnb[(size_t)row * EMBED_DIM + i] = (bf16)v;
    }
}

/* ------------------------------------------------------------------ */
/* S6 selective-scan: one thread per (b,d) channel, serial over L.     */
__global__ __launch_bounds__(256)
void s6_scan_kernel(const float* __restrict__ xn, const float* __restrict__ dtb,
                    const float* __restrict__ xdbl,
                    const float* __restrict__ A_log, const float* __restrict__ D_p,
                    bf16* __restrict__ mxb)
{
    const int ch = blockIdx.x * 256 + threadIdx.x;   /* 0..2047 */
    const int b = ch >> 10, d = ch & 1023;
    float Ac[DSTATE], h[DSTATE];
    for (int n = 0; n < DSTATE; ++n) { Ac[n] = -__expf(A_log[d * DSTATE + n]); h[n] = 0.f; }
    const float Dp = D_p[d];
    for (int l = 0; l < SEQ; ++l) {
        const size_t row = (size_t)b * SEQ + l;
        const float dt = dtb[row * EMBED_DIM + d];
        const float u  = xn [row * EMBED_DIM + d];
        const float* bc = xdbl + row * XDBL_N;
        const float du = dt * u;
        float y = 0.f;
#pragma unroll
        for (int n = 0; n < DSTATE; ++n) {
            float hn = __expf(dt * Ac[n]) * h[n] + du * bc[DT_RANK + n];
            h[n] = hn;
            y += hn * bc[DT_RANK + DSTATE + n];
        }
        float val = y + u * Dp;
        mxb[row * EMBED_DIM + d] = (bf16)(val / (1.f + __expf(-val)));   /* silu */
    }
}

/* ------------------------------------------------------------------ */
/* Build Q' = [q*z^-1/2 | rot(alpha,m)], K' = [k | rot(beta,m)] bf16.  */
/* Folds rel_pos_bias into the qk GEMM: bias[m,n] = a(m).b(n).         */
__global__ __launch_bounds__(128)
void qk_build_kernel(const float* __restrict__ z, const float* __restrict__ gamma,
                     const float* __restrict__ beta, const float* __restrict__ alpha_rb,
                     const float* __restrict__ beta_rb,
                     bf16* __restrict__ Qp, bf16* __restrict__ Kp)
{
    const int row = blockIdx.x;                 /* 0..8191 */
    const int m   = (row % SEQ) % CHUNK;        /* position in chunk */
    const int j   = threadIdx.x;                /* 0..127 */
    const float zv = z[(size_t)row * ZDIM + j];
    float q = zv * gamma[j]        + beta[j];
    float k = zv * gamma[ZDIM + j] + beta[ZDIM + j];
    q *= 0.08838834764831845f;                  /* ZDIM^-0.5 */
    const int half = ZDIM / 2;
    const int jj = (j < half) ? j : (j - half);
    const float f = __expf(-(logf(10000.f) / (float)half) * (float)jj);
    const float ang = (float)m * f;
    const float s = __sinf(ang), c = __cosf(ang);
    const float a1 = alpha_rb[jj], a2 = alpha_rb[jj + half];
    const float b1 = beta_rb[jj],  b2 = beta_rb[jj + half];
    const float ar = (j < half) ? (a1 * c - a2 * s) : (a2 * c + a1 * s);
    const float br = (j < half) ? (b1 * c - b2 * s) : (b2 * c + b1 * s);
    bf16* Q  = Qp + (size_t)row * (2 * ZDIM);
    bf16* Kr = Kp + (size_t)row * (2 * ZDIM);
    Q[j]  = (bf16)q;  Q[ZDIM + j]  = (bf16)ar;
    Kr[j] = (bf16)k;  Kr[ZDIM + j] = (bf16)br;
}

/* ------------------------------------------------------------------ */
/* Exact causal softmax over one chunk row (2048 cols), write bf16.    */
__global__ __launch_bounds__(256)
void softmax_kernel(const float* __restrict__ qk, bf16* __restrict__ attn)
{
    __shared__ float red[256];
    const int r = blockIdx.x;                   /* 0..2047 */
    const float* qr = qk + (size_t)r * CHUNK;
    float vals[8];
    float mx = -1e30f;
    for (int i = 0; i < 8; ++i) {
        int col = threadIdx.x + i * 256;
        float v = (col <= r) ? qr[col] : -INFINITY;
        vals[i] = v; mx = fmaxf(mx, v);
    }
    red[threadIdx.x] = mx; __syncthreads();
    for (int s = 128; s > 0; s >>= 1) {
        if (threadIdx.x < s) red[threadIdx.x] = fmaxf(red[threadIdx.x], red[threadIdx.x + s]);
        __syncthreads();
    }
    mx = red[0]; __syncthreads();
    float sum = 0.f;
    for (int i = 0; i < 8; ++i) {
        float e = (vals[i] == -INFINITY) ? 0.f : __expf(vals[i] - mx);
        vals[i] = e; sum += e;
    }
    red[threadIdx.x] = sum; __syncthreads();
    for (int s = 128; s > 0; s >>= 1) {
        if (threadIdx.x < s) red[threadIdx.x] += red[threadIdx.x + s];
        __syncthreads();
    }
    const float inv = 1.f / red[0];
    for (int i = 0; i < 8; ++i) {
        int col = threadIdx.x + i * 256;
        attn[(size_t)r * CHUNK + col] = (bf16)(vals[i] * inv);
    }
}

/* ------------------------------------------------------------------ */
__global__ void mul_to_bf16_kernel(const float* __restrict__ a,
                                   const float* __restrict__ b,
                                   bf16* __restrict__ o, int n)
{
    int i = blockIdx.x * blockDim.x + threadIdx.x;
    if (i < n) o[i] = (bf16)(a[i] * b[i]);
}

__global__ void final_kernel(const float* __restrict__ x, const float* __restrict__ u,
                             const float* __restrict__ hx, const float* __restrict__ g,
                             float* __restrict__ out, int n)
{
    int i = blockIdx.x * blockDim.x + threadIdx.x;
    if (i < n) {
        float h2 = hx[i] + g[i];
        h2 = h2 / (1.f + __expf(-h2));          /* silu */
        float xi = x[i];
        out[i] = xi + u[i] * (h2 - xi);
    }
}

/* ------------------------------------------------------------------ */
extern "C" void kernel_launch(void* const* d_in, const int* in_sizes, int n_in,
                              void* d_out, int out_size, void* d_ws, size_t ws_size,
                              hipStream_t stream)
{
    const float* x        = (const float*)d_in[0];
    const float* norm_w   = (const float*)d_in[1];
    const float* v_w      = (const float*)d_in[2];
    const float* v_b      = (const float*)d_in[3];
    const float* mx_w     = (const float*)d_in[4];
    const float* mx_b     = (const float*)d_in[5];
    const float* h_w      = (const float*)d_in[6];
    const float* h_b      = (const float*)d_in[7];
    const float* gamma    = (const float*)d_in[8];
    const float* beta     = (const float*)d_in[9];
    const float* alpha_rb = (const float*)d_in[10];
    const float* beta_rb  = (const float*)d_in[11];
    const float* xproj_w  = (const float*)d_in[12];
    const float* dt_w     = (const float*)d_in[13];
    const float* dt_b     = (const float*)d_in[14];
    const float* A_log    = (const float*)d_in[15];
    const float* D_p      = (const float*)d_in[16];
    float* out = (float*)d_out;

    char* p = (char*)d_ws;
    auto alloc = [&](size_t bytes) -> void* {
        void* r = (void*)p;
        p += (bytes + 255) & ~(size_t)255;
        return r;
    };
    const size_t RD = (size_t)ROWS * EMBED_DIM;
    float* xn32   = (float*)alloc(RD * 4);
    bf16*  xnb    = (bf16*) alloc(RD * 2);
    bf16*  vb     = (bf16*) alloc(RD * 2);
    float* xdbl32 = (float*)alloc((size_t)ROWS * XDBL_N * 4);
    bf16*  xdblb  = (bf16*) alloc((size_t)ROWS * XDBL_N * 2);
    float* dt32   = (float*)alloc(RD * 4);
    bf16*  mxb    = (bf16*) alloc(RD * 2);
    float* u32    = (float*)alloc(RD * 4);
    float* z32    = (float*)alloc((size_t)ROWS * ZDIM * 4);
    float* r32    = (float*)alloc(RD * 4);
    float* hx32   = (float*)alloc(RD * 4);
    bf16*  Qp     = (bf16*) alloc((size_t)ROWS * 2 * ZDIM * 2);
    bf16*  Kp     = (bf16*) alloc((size_t)ROWS * 2 * ZDIM * 2);
    float* qk32   = (float*)alloc((size_t)CHUNK * CHUNK * 4);
    bf16*  attnb  = (bf16*) alloc((size_t)CHUNK * CHUNK * 2);
    float* h32    = (float*)alloc(RD * 4);
    bf16*  hrb    = (bf16*) alloc(RD * 2);
    float* g32    = (float*)alloc(RD * 4);

    auto ggrid = [](int M, int N) { return dim3((N + BN - 1) / BN, (M + BM - 1) / BM); };

    /* 1) rmsnorm */
    rmsnorm_kernel<<<ROWS, 256, 0, stream>>>(x, norm_w, xn32, xnb);

    /* 2) v = silu(xn @ v_w^T + v_b)  -> bf16 (attention values) */
    gemm_wmma<0,0,1><<<ggrid(ROWS, EMBED_DIM), 256, 0, stream>>>(
        xnb, EMBED_DIM, v_w, EMBED_DIM, v_b,
        nullptr, vb, EMBED_DIM, ROWS, EMBED_DIM, EMBED_DIM);

    /* 3) x_dbl = xn @ xproj_w^T  (N=96) */
    gemm_wmma<0,0,0><<<ggrid(ROWS, XDBL_N), 256, 0, stream>>>(
        xnb, EMBED_DIM, xproj_w, EMBED_DIM, nullptr,
        xdbl32, xdblb, XDBL_N, ROWS, XDBL_N, EMBED_DIM);

    /* 4) dt = softplus(x_dbl[:, :64] @ dt_w^T + dt_b) */
    gemm_wmma<0,0,3><<<ggrid(ROWS, EMBED_DIM), 256, 0, stream>>>(
        xdblb, XDBL_N, dt_w, DT_RANK, dt_b,
        dt32, nullptr, EMBED_DIM, ROWS, EMBED_DIM, DT_RANK);

    /* 5) S6 scan + D skip + silu -> mx (bf16) */
    s6_scan_kernel<<<(BATCH * EMBED_DIM) / 256, 256, 0, stream>>>(
        xn32, dt32, xdbl32, A_log, D_p, mxb);

    /* 6) base = mx @ mx_w^T split into four fused GEMMs */
    gemm_wmma<0,0,2><<<ggrid(ROWS, EMBED_DIM), 256, 0, stream>>>(         /* u: sigmoid */
        mxb, EMBED_DIM, mx_w, EMBED_DIM, mx_b,
        u32, nullptr, EMBED_DIM, ROWS, EMBED_DIM, EMBED_DIM);
    gemm_wmma<0,0,1><<<ggrid(ROWS, ZDIM), 256, 0, stream>>>(              /* z: silu */
        mxb, EMBED_DIM, mx_w + (size_t)EMBED_DIM * EMBED_DIM, EMBED_DIM,
        mx_b + EMBED_DIM, z32, nullptr, ZDIM, ROWS, ZDIM, EMBED_DIM);
    gemm_wmma<0,0,1><<<ggrid(ROWS, EMBED_DIM), 256, 0, stream>>>(         /* r: silu */
        mxb, EMBED_DIM, mx_w + (size_t)(EMBED_DIM + ZDIM) * EMBED_DIM, EMBED_DIM,
        mx_b + EMBED_DIM + ZDIM, r32, nullptr, EMBED_DIM, ROWS, EMBED_DIM, EMBED_DIM);
    gemm_wmma<0,0,0><<<ggrid(ROWS, EMBED_DIM), 256, 0, stream>>>(         /* hx: linear */
        mxb, EMBED_DIM, mx_w + (size_t)(2 * EMBED_DIM + ZDIM) * EMBED_DIM, EMBED_DIM,
        mx_b + 2 * EMBED_DIM + ZDIM, hx32, nullptr, EMBED_DIM, ROWS, EMBED_DIM, EMBED_DIM);

    /* 7) Q'/K' with folded rotary rel-pos bias */
    qk_build_kernel<<<ROWS, 128, 0, stream>>>(z32, gamma, beta, alpha_rb, beta_rb, Qp, Kp);

    /* 8) chunked attention: qk GEMM -> masked softmax -> attn@v GEMM */
    for (int c = 0; c < (ROWS / CHUNK); ++c) {
        const size_t rowbase = (size_t)c * CHUNK;
        gemm_wmma<1,0,0><<<ggrid(CHUNK, CHUNK), 256, 0, stream>>>(
            Qp + rowbase * 2 * ZDIM, 2 * ZDIM,
            Kp + rowbase * 2 * ZDIM, 2 * ZDIM, nullptr,
            qk32, nullptr, CHUNK, CHUNK, CHUNK, 2 * ZDIM);
        softmax_kernel<<<CHUNK, 256, 0, stream>>>(qk32, attnb);
        gemm_wmma<1,1,0><<<ggrid(CHUNK, EMBED_DIM), 256, 0, stream>>>(
            attnb, CHUNK,
            vb + rowbase * EMBED_DIM, EMBED_DIM, nullptr,
            h32 + rowbase * EMBED_DIM, nullptr, EMBED_DIM,
            CHUNK, EMBED_DIM, CHUNK);
    }

    /* 9) hr = h * r (bf16), then g = hr @ h_w^T + h_b */
    const int NTOT = ROWS * EMBED_DIM;
    mul_to_bf16_kernel<<<(NTOT + 255) / 256, 256, 0, stream>>>(h32, r32, hrb, NTOT);
    gemm_wmma<0,0,0><<<ggrid(ROWS, EMBED_DIM), 256, 0, stream>>>(
        hrb, EMBED_DIM, h_w, EMBED_DIM, h_b,
        g32, nullptr, EMBED_DIM, ROWS, EMBED_DIM, EMBED_DIM);

    /* 10) out = x + u * (silu(hx + g) - x) */
    final_kernel<<<(NTOT + 255) / 256, 256, 0, stream>>>(x, u32, hx32, g32, out, NTOT);
}
